// Dconv_cos_21827023798971
// MI455X (gfx1250) — compile-verified, matrix operand order checked by reading, not run on hardware
//
#include <hip/hip_runtime.h>

// ---------------------------------------------------------------------------
// Problem constants (fixed by the reference setup)
// ---------------------------------------------------------------------------
#define BATCH 32
#define CH    64
#define HH    56
#define WW    56
#define PP    (HH * WW)          // 3136 pixels per image
#define OC    64
#define KDIM  (CH * 9)           // 576 = GEMM K (k = rs*64 + c)
#define KPAD  584                // padded K stride (halves); 1168 B row, 16B ok
#define TN    32                 // pixels (N) per workgroup in conv kernel

#define TILE  16
#define HALO  3
#define TW    (TILE + 2 * HALO)  // 22 halo tile side
#define LSTR  68                 // padded float stride per halo pixel (16B ok)

typedef __attribute__((ext_vector_type(16))) _Float16 v16h;
typedef __attribute__((ext_vector_type(8)))  float    v8f;

// ---------------------------------------------------------------------------
// Kernel A: per-pixel L2 norm + transpose.
//   x[B,C,P] (f32)  ->  xn_t[B,P,C] (f32, normalized), xt_h[B,P,C] (f16, raw)
// ---------------------------------------------------------------------------
__global__ void Dcos_norm_transpose(const float* __restrict__ x,
                                    float* __restrict__ xn_t,
                                    _Float16* __restrict__ xt_h) {
  int gid = blockIdx.x * blockDim.x + threadIdx.x;   // b*P + p  (exact grid)
  int b = gid / PP;
  int p = gid - b * PP;
  const float* xp = x + (size_t)b * CH * PP + p;
  float v[CH];
  float ss = 0.0f;
#pragma unroll
  for (int c = 0; c < CH; ++c) {
    v[c] = xp[(size_t)c * PP];                       // lanes: coalesced over p
    ss = fmaf(v[c], v[c], ss);
  }
  float inv = 1.0f / fmaxf(sqrtf(ss), 1e-6f);
  float*    od = xn_t + (size_t)gid * CH;
  _Float16* oh = xt_h + (size_t)gid * CH;
#pragma unroll
  for (int c = 0; c < CH; ++c) {
    od[c] = v[c] * inv;
    oh[c] = (_Float16)v[c];
  }
}

// ---------------------------------------------------------------------------
// Kernel B: LDS-tiled 7x7 cosine correlation + stable top-9 + index sort.
//   xn_t[B,P,C] -> cand[B,P,9] (sorted ascending neighbor 1D indices)
// ---------------------------------------------------------------------------
__global__ void Dcos_topk(const float* __restrict__ xn_t,
                          int* __restrict__ cand) {
  extern __shared__ float lds[];                     // TW*TW*LSTR floats
  const int t = threadIdx.x;                         // 256 threads = 8 waves
  const int b = blockIdx.z;
  const int oi = blockIdx.y * TILE - HALO;
  const int oj = blockIdx.x * TILE - HALO;
  const float* xb = xn_t + (size_t)b * PP * CH;

  // Stage 22x22 halo of normalized vectors into LDS (float4 granularity).
  for (int idx = t; idx < TW * TW * (CH / 4); idx += 256) {
    int pix = idx >> 4, c4 = idx & 15;
    int gi = oi + pix / TW, gj = oj + pix % TW;
    float4 val = make_float4(0.f, 0.f, 0.f, 0.f);
    if (gi >= 0 && gi < HH && gj >= 0 && gj < WW)
      val = *(const float4*)(xb + (size_t)(gi * WW + gj) * CH + c4 * 4);
    *(float4*)(lds + pix * LSTR + c4 * 4) = val;
  }
  __syncthreads();

  const int ti = t >> 4, tj = t & 15;
  const int gi = blockIdx.y * TILE + ti;
  const int gj = blockIdx.x * TILE + tj;
  if (gi >= HH || gj >= WW) return;                  // edge tiles partial

  const float* own = lds + ((ti + HALO) * TW + (tj + HALO)) * LSTR;
  float4 ox[CH / 4];
#pragma unroll
  for (int c4 = 0; c4 < CH / 4; ++c4) ox[c4] = *(const float4*)(own + c4 * 4);

  float cosv[9];
  int   idxv[9];
#pragma unroll
  for (int s = 0; s < 9; ++s) { cosv[s] = -3.0f; idxv[s] = 0; }

  // Enumerate candidates in the reference's row-major (di,dj) order.
  for (int di = -HALO; di <= HALO; ++di) {
    for (int dj = -HALO; dj <= HALO; ++dj) {
      int ni = gi + di, nj = gj + dj;
      if (ni < 0 || ni >= HH || nj < 0 || nj >= WW) continue;  // masked: skip
      const float* nb = lds + ((ti + HALO + di) * TW + (tj + HALO + dj)) * LSTR;
      float acc = 0.0f;
#pragma unroll
      for (int c4 = 0; c4 < CH / 4; ++c4) {
        float4 bb = *(const float4*)(nb + c4 * 4);
        acc = fmaf(ox[c4].x, bb.x, acc);
        acc = fmaf(ox[c4].y, bb.y, acc);
        acc = fmaf(ox[c4].z, bb.z, acc);
        acc = fmaf(ox[c4].w, bb.w, acc);
      }
      // Stable branch-free insertion into the 9-deep descending list.
      float cc = acc;
      int qq = ni * WW + nj;
#pragma unroll
      for (int s = 0; s < 9; ++s) {
        bool sw = cc > cosv[s];
        float tc = cosv[s]; int tq = idxv[s];
        cosv[s] = sw ? cc : tc;  idxv[s] = sw ? qq : tq;
        cc = sw ? tc : cc;       qq = sw ? tq : qq;
      }
    }
  }

  // Sort winner indices ascending (bubble network, fully unrolled).
#pragma unroll
  for (int a = 0; a < 8; ++a)
#pragma unroll
    for (int s = 0; s < 8; ++s) {
      int lo = idxv[s] < idxv[s + 1] ? idxv[s] : idxv[s + 1];
      int hi = idxv[s] < idxv[s + 1] ? idxv[s + 1] : idxv[s];
      idxv[s] = lo; idxv[s + 1] = hi;
    }

  int* dst = cand + ((size_t)b * PP + gi * WW + gj) * 9;
#pragma unroll
  for (int s = 0; s < 9; ++s) dst[s] = idxv[s];
}

// ---------------------------------------------------------------------------
// Kernel C: gathered GEMM  out[64, B*P] = W[64,576] x A[576, B*P]  via WMMA.
//   Per workgroup: M=64 (full O), N=32 pixels, K=576, f16 in / f32 acc.
//   8 waves: wave w -> (mrow = w&3, ncol = w>>2), one 16x16 D tile each.
// ---------------------------------------------------------------------------
__global__ void Dcos_conv_wmma(const _Float16* __restrict__ xt_h,
                               const float* __restrict__ conv_w,
                               const int* __restrict__ cand,
                               float* __restrict__ out) {
  extern __shared__ char smem[];
  _Float16* Wlds  = (_Float16*)smem;                             // [OC][KPAD]
  _Float16* Blds  = (_Float16*)(smem + OC * KPAD * 2);           // [TN][KPAD]
  int*      cands = (int*)(smem + OC * KPAD * 2 + TN * KPAD * 2);// [TN*9]

  const int t = threadIdx.x;                                     // 256 = 8 waves
  const int tilesPerImg = PP / TN;                               // 98
  const int b = blockIdx.x / tilesPerImg;
  const int tile = blockIdx.x - b * tilesPerImg;
  const int p0 = tile * TN;

  // Phase 1: weights (f32 -> f16, remap k = rs*64 + c), division-free:
  // thread owns column k, walks all 64 output channels (stride 576 floats).
  for (int k = t; k < KDIM; k += 256) {
    int rs = k >> 6, c = k & 63;
    const float* wp = conv_w + c * 9 + rs;
    _Float16* wl = Wlds + k;
#pragma unroll
    for (int o = 0; o < OC; ++o)
      wl[o * KPAD] = (_Float16)wp[o * KDIM];
  }
  for (int idx = t; idx < TN * 9; idx += 256)
    cands[idx] = cand[((size_t)b * PP + p0) * 9 + idx];
  __syncthreads();

  // Phase 2: gather B tile. One thread per (pixel n, slot rs): 128B row copy.
  for (int chunk = t; chunk < TN * 9; chunk += 256) {
    int n = chunk / 9, rs = chunk - n * 9;
    int q = cands[chunk];
    const uint4* src = (const uint4*)(xt_h + ((size_t)b * PP + q) * CH);
    uint4* dst = (uint4*)(Blds + n * KPAD + rs * 64);
#pragma unroll
    for (int j = 0; j < 8; ++j) dst[j] = src[j];
  }
  __syncthreads();

  // Phase 3: WMMA main loop, fully unrolled (18 x v_wmma_f32_16x16x32_f16).
  // ISA 16-bit 16x32 operand layout: lanes 0-15 hold K 0-7 (v0-3) and
  // 16-23 (v4-7); lanes 16-31 hold K 8-15 and 24-31. Each half-fragment is a
  // contiguous, 16B-aligned run of 8 halves -> one ds_load_b128 each.
  const int wid = t >> 5, lane = t & 31;
  const int mrow = wid & 3, ncol = wid >> 2;
  const int l15 = lane & 15, hi8 = (lane >> 4) * 8;
  const _Float16* Arow = Wlds + (mrow * 16 + l15) * KPAD + hi8;
  const _Float16* Brow = Blds + (ncol * 16 + l15) * KPAD + hi8;

  union Frag { v16h h; uint4 q[2]; };
  v8f acc = {};
#pragma unroll
  for (int kk = 0; kk < KDIM; kk += 32) {
    Frag a, bf;
    a.q[0]  = *(const uint4*)(Arow + kk);        // halves kk+hi8   .. +7
    a.q[1]  = *(const uint4*)(Arow + kk + 16);   // halves kk+16+hi8.. +7
    bf.q[0] = *(const uint4*)(Brow + kk);
    bf.q[1] = *(const uint4*)(Brow + kk + 16);
    acc = __builtin_amdgcn_wmma_f32_16x16x32_f16(
        /*neg_a=*/false, a.h, /*neg_b=*/false, bf.h,
        /*c_mod=*/(short)0, acc, /*reuse_a=*/false, /*reuse_b=*/false);
  }

  // Phase 4: scatter D per C/D layout: VGPR r -> M = r + 8*hi, N = l15.
  float* ob = out + (size_t)b * OC * PP + (p0 + ncol * 16 + l15);
#pragma unroll
  for (int r = 0; r < 8; ++r) {
    int o = mrow * 16 + r + hi8 / 8 * 8;         // == mrow*16 + r + 8*hi
    ob[(size_t)o * PP] = acc[r];
  }
}

// ---------------------------------------------------------------------------
// Launch
// ---------------------------------------------------------------------------
extern "C" void kernel_launch(void* const* d_in, const int* in_sizes, int n_in,
                              void* d_out, int out_size, void* d_ws, size_t ws_size,
                              hipStream_t stream) {
  (void)in_sizes; (void)n_in; (void)out_size; (void)ws_size;
  const float* x      = (const float*)d_in[0];
  const float* conv_w = (const float*)d_in[1];
  // d_in[2] (nbr_idx) / d_in[3] (nbr_mask) are recomputed on the fly.
  float* out = (float*)d_out;

  char* ws = (char*)d_ws;
  float*    xn_t = (float*)ws;                               // 25,690,112 B
  _Float16* xt_h = (_Float16*)(ws + 25690112);               // 12,845,056 B
  int*      cand = (int*)(ws + 38535168);                    //  3,612,672 B

  // A: normalize + transpose. Exactly B*P threads.
  Dcos_norm_transpose<<<(BATCH * PP) / 256, 256, 0, stream>>>(x, xn_t, xt_h);

  // B: tiled cosine + top-9. 4x4 tiles per image, halo tile in dynamic LDS.
  dim3 gB((WW + TILE - 1) / TILE, (HH + TILE - 1) / TILE, BATCH);
  size_t ldsB = (size_t)TW * TW * LSTR * sizeof(float);      // 131,648 B
  Dcos_topk<<<gB, 256, ldsB, stream>>>(xn_t, cand);

  // C: gathered GEMM via WMMA. One block per (b, 32-pixel tile).
  size_t ldsC = (size_t)OC * KPAD * 2 + (size_t)TN * KPAD * 2 + TN * 9 * 4;
  Dcos_conv_wmma<<<BATCH * (PP / TN), 256, ldsC, stream>>>(xt_h, conv_w, cand, out);
}
